// Block_45947560133327
// MI455X (gfx1250) — matrix operand order repaired and checked by reading, not compiled, not run on hardware
//
#include <hip/hip_runtime.h>
#include <hip/hip_bf16.h>

// ---------------------------------------------------------------------------
// Divided space-time transformer block on gfx1250 (wave32, WMMA f16->f32,
// TDM tensor_load_to_lds staging for the GEMMs when available).
// ---------------------------------------------------------------------------

typedef __attribute__((ext_vector_type(16))) _Float16 v16h;
typedef __attribute__((ext_vector_type(8)))  _Float16 v8h;
typedef __attribute__((ext_vector_type(8)))  float    v8f;
typedef __attribute__((ext_vector_type(4)))  unsigned int v4u;
typedef __attribute__((ext_vector_type(8)))  int      v8i;
typedef __attribute__((ext_vector_type(4)))  int      v4i;

#define DIMC   512
#define HEADS  8
#define HD     64
#define BB     4
#define TT     64
#define NN     196
#define MTOK   (BB*TT*NN)      // 50176 tokens
#define HIDDEN 2048
#define SCALE  0.125f          // 64^-0.5
#define EPSLN  1e-5f

#if defined(__gfx1250__)
#if __has_builtin(__builtin_amdgcn_tensor_load_to_lds) && \
    __has_builtin(__builtin_amdgcn_s_wait_tensorcnt)
#define USE_TDM 1
#endif
#endif

union FragU { v16h f; v8h h[2]; };

#ifdef USE_TDM
// Raw LDS byte offset of a __shared__ object (addrspacecast -> ptrtoint).
__device__ __forceinline__ unsigned lds_off(const void* p) {
    return (unsigned)(unsigned long long)
        (__attribute__((address_space(3))) const void*)p;
}

// Issue one TDM 2D tile load: tile_d0 elems/row (f16), tile_d1 rows,
// row stride = stride elems, from gptr into LDS at lds_addr.
// D# bitfields per CDNA5 ISA 8.3/8.4 (group0 128b, group1 256b).
__device__ __forceinline__ void tdm_load_2d_f16(unsigned lds_addr,
                                                const void* gptr,
                                                unsigned tile_d0,
                                                unsigned tile_d1,
                                                unsigned long long stride,
                                                unsigned tensor_d0) {
    unsigned long long ga = (unsigned long long)gptr;
    v4u g0;
    g0[0] = 1u;                                    // count=1, user mode
    g0[1] = lds_addr;                              // lds_addr [63:32]
    g0[2] = (unsigned)ga;                          // global_addr lo
    g0[3] = (unsigned)((ga >> 32) & 0x01FFFFFFu)   // global_addr [56:32]
            | (2u << 30);                          // type=2 ("image")
    v8i g1;
    g1[0] = (int)(1u << 16);                       // data_size=1 (2 bytes)
    g1[1] = (int)((tensor_d0 & 0xFFFFu) << 16);    // tensor_dim0 [79:48] lo16
    g1[2] = (int)(((tensor_d0 >> 16) & 0xFFFFu)    // tensor_dim0 hi16
            | ((tile_d1 & 0xFFFFu) << 16));        // tensor_dim1 lo16 = rows
    g1[3] = (int)((tile_d0 & 0xFFFFu) << 16);      // tensor_dim1 hi=0, tile_dim0
    g1[4] = (int)(tile_d1 & 0xFFFFu);              // tile_dim1, tile_dim2=0
    g1[5] = (int)(unsigned)(stride & 0xFFFFFFFFu); // tensor_dim0_stride lo32
    g1[6] = (int)(unsigned)((stride >> 32) & 0xFFFFu); // stride hi16, d1_stride lo=0
    g1[7] = 0;                                     // tensor_dim1_stride hi
    v4i z4 = {0, 0, 0, 0};
#if __has_include(<hip/amd_detail/amd_gfx1250_TDM.h>)
    v8i z8 = {0, 0, 0, 0, 0, 0, 0, 0};
    __builtin_amdgcn_tensor_load_to_lds(g0, g1, z4, z4, z8, 0);
#else
    __builtin_amdgcn_tensor_load_to_lds(g0, g1, z4, z4, 0);
#endif
}
#endif  // USE_TDM

// ---------------------------------------------------------------------------
__global__ void cvt_f16_kernel(const float* __restrict__ src,
                               _Float16* __restrict__ dst, int n) {
    int i = blockIdx.x * 256 + threadIdx.x;
    if (i < n) dst[i] = (_Float16)src[i];
}

__global__ void biascat_kernel(const float* __restrict__ qb,
                               const float* __restrict__ vb,
                               float* __restrict__ out) {
    int i = blockIdx.x * 256 + threadIdx.x;   // 1536 total
    out[i] = (i < 512) ? qb[i] : (i < 1024) ? 0.0f : vb[i - 1024];
}

// ---------------------------------------------------------------------------
// LayerNorm over last dim (512), one block (256 threads) per row, f16 out.
// ---------------------------------------------------------------------------
__global__ void ln_kernel(const float* __restrict__ x,
                          const float* __restrict__ g,
                          const float* __restrict__ b,
                          _Float16* __restrict__ out) {
    __shared__ float red[256];
    int row = blockIdx.x;
    int tid = threadIdx.x;
    const float* xr = x + (size_t)row * DIMC;
    float a0 = xr[tid], a1 = xr[tid + 256];
    red[tid] = a0 + a1;
    __syncthreads();
    for (int o = 128; o > 0; o >>= 1) {
        if (tid < o) red[tid] += red[tid + o];
        __syncthreads();
    }
    float mean = red[0] * (1.0f / DIMC);
    __syncthreads();
    float d0 = a0 - mean, d1 = a1 - mean;
    red[tid] = d0 * d0 + d1 * d1;
    __syncthreads();
    for (int o = 128; o > 0; o >>= 1) {
        if (tid < o) red[tid] += red[tid + o];
        __syncthreads();
    }
    float inv = rsqrtf(red[0] * (1.0f / DIMC) + EPSLN);
    _Float16* orow = out + (size_t)row * DIMC;
    orow[tid]       = (_Float16)(d0 * inv * g[tid] + b[tid]);
    orow[tid + 256] = (_Float16)(d1 * inv * g[tid + 256] + b[tid + 256]);
}

// ---------------------------------------------------------------------------
// Tiled WMMA GEMM: C[M,N] = A[M,K](f16) * W[N,K](f16)^T  (+ epilogue)
// Block = 128 threads (4 waves). Block tile 128x64, K-step 32, double-buffered
// LDS staging with one barrier per K-step. Staging is done by the Tensor Data
// Mover (one 2D-tile DMA per operand, issued by wave 0, TENSORcnt-tracked)
// when the TDM builtin exists; otherwise by vector loads + ds_store.
// Wave w -> rows [w*32, w*32+32) as two 16-row bands, 4 col tiles of 16.
// MODE 0: out f16, += bias (bias = precomputed qkv concat)
// MODE 1: out f32, out = addsrc + 0.5*(acc+bias)
// MODE 2: out f16, out = gelu(acc+bias)
// MODE 3: out f32, out = addsrc + acc + bias
// ---------------------------------------------------------------------------
template <int MODE>
__global__ void gemm_kernel(const _Float16* __restrict__ A,
                            const _Float16* __restrict__ W,
                            const float* __restrict__ bias,
                            const float* __restrict__ addsrc,
                            void* __restrict__ outp,
                            int M, int N, int K) {
    __shared__ _Float16 As[2][128 * 32];   // [row][k]
    __shared__ _Float16 Bs[2][64 * 32];    // [ncol][k]
    const int mbase = blockIdx.x * 128;
    const int nbase = blockIdx.y * 64;
    const int tid   = threadIdx.x;
    const int w     = tid >> 5;
    const int lane  = tid & 31;

    v8f acc[2][4] = {};

#ifdef USE_TDM
    const _Float16* aTile = A + (size_t)mbase * K;
    const _Float16* bTile = W + (size_t)nbase * K;
    auto stage = [&](int buf, int k0) {
        if (w == 0) {
            tdm_load_2d_f16(lds_off(&As[buf][0]), aTile + k0, 32, 128,
                            (unsigned long long)K, (unsigned)K);
            tdm_load_2d_f16(lds_off(&Bs[buf][0]), bTile + k0, 32, 64,
                            (unsigned long long)K, (unsigned)K);
        }
    };
    auto stage_wait = [&]() {
        if (w == 0) __builtin_amdgcn_s_wait_tensorcnt(0);
    };
#else
    // manual staging: A row = tid (32 halfs); B row = tid>>1, half = tid&1
    const _Float16* aSrc = A + (size_t)(mbase + tid) * K;
    const _Float16* bSrc = W + (size_t)(nbase + (tid >> 1)) * K + (tid & 1) * 16;
    _Float16* aDst = As[0] + tid * 32;
    _Float16* bDst = Bs[0] + (tid >> 1) * 32 + (tid & 1) * 16;
    const size_t bufStepA = 128 * 32, bufStepB = 64 * 32;
    auto stage = [&](int buf, int k0) {
        const v8h* ap = (const v8h*)(aSrc + k0);
        v8h* ad = (v8h*)(aDst + buf * bufStepA);
        ad[0] = ap[0]; ad[1] = ap[1]; ad[2] = ap[2]; ad[3] = ap[3];
        const v8h* bp = (const v8h*)(bSrc + k0);
        v8h* bd = (v8h*)(bDst + buf * bufStepB);
        bd[0] = bp[0]; bd[1] = bp[1];
        if (k0 + 32 < K) {
            __builtin_prefetch(aSrc + k0 + 32, 0, 0);
            __builtin_prefetch(bSrc + k0 + 32, 0, 0);
        }
    };
    auto stage_wait = [&]() {};
#endif

    stage(0, 0);
    stage_wait();
    __syncthreads();

    const int hi8   = (lane < 16) ? 0 : 8;    // A-frag run offset
    const int off16 = (lane < 16) ? 0 : 16;   // unused; B frag also uses hi8
    int buf = 0;
    for (int k0 = 0; k0 < K; k0 += 32) {
        if (k0 + 32 < K) stage(buf ^ 1, k0 + 32);
        v16h af[2];
#pragma unroll
        for (int i = 0; i < 2; ++i) {
            const _Float16* p =
                As[buf] + (w * 32 + i * 16 + (lane & 15)) * 32 + hi8;
            FragU u;
            u.h[0] = *(const v8h*)p;
            u.h[1] = *(const v8h*)(p + 16);
            af[i] = u.f;
        }
#pragma unroll
        for (int j = 0; j < 4; ++j) {
            const _Float16* p = Bs[buf] + (j * 16 + (lane & 15)) * 32 + hi8;
            FragU u;
            u.h[0] = *(const v8h*)p;
            u.h[1] = *(const v8h*)(p + 16);
#pragma unroll
            for (int i = 0; i < 2; ++i)
                acc[i][j] = __builtin_amdgcn_wmma_f32_16x16x32_f16(
                    false, af[i], false, u.f, (short)0, acc[i][j], false, false);
        }
        stage_wait();
        __syncthreads();
        buf ^= 1;
    }

    // epilogue
#pragma unroll
    for (int i = 0; i < 2; ++i) {
#pragma unroll
        for (int j = 0; j < 4; ++j) {
#pragma unroll
            for (int r = 0; r < 8; ++r) {
                const int grow =
                    mbase + w * 32 + i * 16 + r + ((lane < 16) ? 0 : 8);
                const int gcol = nbase + j * 16 + (lane & 15);
                const size_t idx = (size_t)grow * N + gcol;
                float v = acc[i][j][r];
                if (MODE == 0) {
                    ((_Float16*)outp)[idx] = (_Float16)(v + bias[gcol]);
                } else if (MODE == 1) {
                    ((float*)outp)[idx] = addsrc[idx] + 0.5f * (v + bias[gcol]);
                } else if (MODE == 2) {
                    float t = v + bias[gcol];
                    float ge =
                        0.5f * t * (1.0f + erff(t * 0.70710678118654752f));
                    ((_Float16*)outp)[idx] = (_Float16)ge;
                } else {
                    ((float*)outp)[idx] = addsrc[idx] + v + bias[gcol];
                }
            }
        }
    }
}

// ---------------------------------------------------------------------------
// Multi-head attention over a strided token axis (WMMA).
// qkv layout: [token][1536] = q(512) | k(512) | v(512); head h at h*64.
// Block = 64 threads (2 waves); one block per (instance, head).
// Q/K fragments are read straight from global as 16B vector loads;
// V is staged transposed in LDS; scores live in per-wave fp32 LDS rows.
// ---------------------------------------------------------------------------
template <int S, int SP, bool TEMPORAL>
__global__ void attn_kernel(const _Float16* __restrict__ qkv,
                            _Float16* __restrict__ aout,
                            const float* __restrict__ mask) {
    constexpr int NB = (S + 15) / 16;
    __shared__ _Float16 Vt[64 * SP];        // [d][pos], padded pos -> 0
    __shared__ float    Sc[2 * 16 * SP];    // per-wave score rows

    const int inst = blockIdx.x;
    const int h    = inst & 7;
    const int g    = inst >> 3;
    int base, stride;
    if (TEMPORAL) {
        const int b = g / NN, n = g % NN;
        base = b * TT * NN + n;  stride = NN;
    } else {
        base = g * NN;           stride = 1;
    }
    const int tid  = threadIdx.x;
    const int w    = tid >> 5;
    const int lane = tid & 31;
    const int hi8   = (lane < 16) ? 0 : 8;
    const int off16 = (lane < 16) ? 0 : 16;

    // stage V transposed into LDS (vector global reads, scatter stores)
    for (int idx = tid; idx < SP * 8; idx += 64) {
        const int pos = idx >> 3;
        const int d8  = (idx & 7) * 8;
        v8h val = {};
        if (pos < S)
            val = *(const v8h*)(qkv + (size_t)(base + stride * pos) * 1536 +
                                1024 + h * 64 + d8);
#pragma unroll
        for (int q = 0; q < 8; ++q) Vt[(d8 + q) * SP + pos] = val[q];
    }
    __syncthreads();

    float* sc = Sc + w * 16 * SP;

    for (int band = w; band < NB; band += 2) {
        // ---- scores: S_tile = Q(16 x 64) * K^T ----
        const int qrow = band * 16 + (lane & 15);
        const bool qok = (qrow < S);
        const _Float16* qp =
            qkv + (size_t)(base + stride * (qok ? qrow : 0)) * 1536 + h * 64;
        v16h aq[2];
#pragma unroll
        for (int t = 0; t < 2; ++t) {
            FragU u;
            u.h[0] = (v8h){};
            u.h[1] = (v8h){};
            if (qok) {
                const _Float16* p = qp + t * 32 + hi8;
                u.h[0] = *(const v8h*)p;
                u.h[1] = *(const v8h*)(p + 16);
            }
            aq[t] = u.f;
        }
#pragma unroll 1
        for (int j = 0; j < SP / 16; ++j) {
            const int kcol = j * 16 + (lane & 15);
            const bool kok = (kcol < S);
            const _Float16* kp =
                qkv + (size_t)(base + stride * (kok ? kcol : 0)) * 1536 + 512 +
                h * 64;
            v8f acc = {};
#pragma unroll
            for (int t = 0; t < 2; ++t) {
                FragU u;
                u.h[0] = (v8h){};
                u.h[1] = (v8h){};
                if (kok) {
                    const _Float16* p = kp + t * 32 + hi8;
                    u.h[0] = *(const v8h*)p;
                    u.h[1] = *(const v8h*)(p + 16);
                }
                acc = __builtin_amdgcn_wmma_f32_16x16x32_f16(
                    false, aq[t], false, u.f, (short)0, acc, false, false);
            }
#pragma unroll
            for (int r = 0; r < 8; ++r)
                sc[(r + ((lane < 16) ? 0 : 8)) * SP + j * 16 + (lane & 15)] =
                    acc[r];
        }

        // ---- row softmax (lane pair r / r+16 split the columns) ----
        {
            const int r = lane & 15;
            const int halfsel = lane >> 4;
            const int rowg = band * 16 + r;
            const int c0 = halfsel * (S / 2);
            const int c1 = halfsel ? S : (S / 2);
            const float* mrow = TEMPORAL ? (mask + rowg * TT) : nullptr;
            float mx = -1e30f;
            for (int c = c0; c < c1; ++c) {
                float v = sc[r * SP + c] * SCALE;
                if (TEMPORAL) v += mrow[c];
                mx = fmaxf(mx, v);
            }
            mx = fmaxf(mx, __shfl_xor(mx, 16, 32));
            float sm = 0.0f;
            for (int c = c0; c < c1; ++c) {
                float v = sc[r * SP + c] * SCALE;
                if (TEMPORAL) v += mrow[c];
                float e_ = __expf(v - mx);
                sm += e_;
                sc[r * SP + c] = e_;
            }
            sm += __shfl_xor(sm, 16, 32);
            const float inv = 1.0f / sm;
            for (int c = c0; c < c1; ++c) sc[r * SP + c] *= inv;
            // cols in [S,SP): stale finite scores * zero V rows -> contribute 0
        }

        // ---- O band = P(16 x SP) * V(SP x 64) ----
#pragma unroll
        for (int jo = 0; jo < 4; ++jo) {
            v8f acc = {};
#pragma unroll 1
            for (int kk = 0; kk < SP; kk += 32) {
                FragU a, b;
                const float* sp_ = sc + (lane & 15) * SP + kk + hi8;
#pragma unroll
                for (int e = 0; e < 8; ++e) a.h[0][e] = (_Float16)sp_[e];
#pragma unroll
                for (int e = 0; e < 8; ++e) a.h[1][e] = (_Float16)sp_[16 + e];
                const _Float16* vp =
                    Vt + (jo * 16 + (lane & 15)) * SP + kk + hi8;
                b.h[0] = *(const v8h*)vp;
                b.h[1] = *(const v8h*)(vp + 16);
                acc = __builtin_amdgcn_wmma_f32_16x16x32_f16(
                    false, a.f, false, b.f, (short)0, acc, false, false);
            }
#pragma unroll
            for (int r = 0; r < 8; ++r) {
                const int rl = r + ((lane < 16) ? 0 : 8);
                const int rg = band * 16 + rl;
                if (rg < S) {
                    const size_t tok = (size_t)(base + stride * rg);
                    aout[tok * 512 + h * 64 + jo * 16 + (lane & 15)] =
                        (_Float16)acc[r];
                }
            }
        }
    }
}

// ---------------------------------------------------------------------------
// Host-side orchestration
// ---------------------------------------------------------------------------
extern "C" void kernel_launch(void* const* d_in, const int* in_sizes, int n_in,
                              void* d_out, int out_size, void* d_ws, size_t ws_size,
                              hipStream_t stream) {
    (void)in_sizes; (void)n_in; (void)out_size; (void)ws_size;
    const float* x      = (const float*)d_in[0];
    const float* mask   = (const float*)d_in[1];
    const float* n1_g   = (const float*)d_in[2];
    const float* n1_b   = (const float*)d_in[3];
    const float* s_qkvw = (const float*)d_in[4];
    const float* s_q_b  = (const float*)d_in[5];
    const float* s_v_b  = (const float*)d_in[6];
    const float* s_pw   = (const float*)d_in[7];
    const float* s_p_b  = (const float*)d_in[8];
    const float* t_qkvw = (const float*)d_in[9];
    const float* t_q_b  = (const float*)d_in[10];
    const float* t_v_b  = (const float*)d_in[11];
    const float* t_pw   = (const float*)d_in[12];
    const float* t_p_b  = (const float*)d_in[13];
    const float* n2_g   = (const float*)d_in[14];
    const float* n2_b   = (const float*)d_in[15];
    const float* w1     = (const float*)d_in[16];
    const float* b1     = (const float*)d_in[17];
    const float* w2     = (const float*)d_in[18];
    const float* b2     = (const float*)d_in[19];

    char* ws = (char*)d_ws;
    size_t off = 0;
    auto alloc = [&](size_t bytes) -> void* {
        void* p = ws + off;
        off += bytes;
        off = (off + 255) & ~(size_t)255;
        return p;
    };
    _Float16* wqs  = (_Float16*)alloc((size_t)1536 * 512 * 2);
    _Float16* wqt  = (_Float16*)alloc((size_t)1536 * 512 * 2);
    _Float16* wps  = (_Float16*)alloc((size_t)512 * 512 * 2);
    _Float16* wpt  = (_Float16*)alloc((size_t)512 * 512 * 2);
    _Float16* w1h  = (_Float16*)alloc((size_t)HIDDEN * 512 * 2);
    _Float16* w2h  = (_Float16*)alloc((size_t)512 * HIDDEN * 2);
    float*    bcs  = (float*)   alloc((size_t)1536 * 4);
    float*    bct  = (float*)   alloc((size_t)1536 * 4);
    _Float16* hbuf = (_Float16*)alloc((size_t)MTOK * 512 * 2);
    float*    xnew = (float*)   alloc((size_t)MTOK * 512 * 4);
    _Float16* qkvb = (_Float16*)alloc((size_t)MTOK * HIDDEN * 2); // qkv/ffn
    _Float16* aob  = (_Float16*)alloc((size_t)MTOK * 512 * 2);

    auto cvt = [&](const float* src, _Float16* dst, int n) {
        cvt_f16_kernel<<<(n + 255) / 256, 256, 0, stream>>>(src, dst, n);
    };
    cvt(s_qkvw, wqs, 1536 * 512);
    cvt(t_qkvw, wqt, 1536 * 512);
    cvt(s_pw,   wps, 512 * 512);
    cvt(t_pw,   wpt, 512 * 512);
    cvt(w1,     w1h, HIDDEN * 512);
    cvt(w2,     w2h, 512 * HIDDEN);
    biascat_kernel<<<6, 256, 0, stream>>>(s_q_b, s_v_b, bcs);
    biascat_kernel<<<6, 256, 0, stream>>>(t_q_b, t_v_b, bct);

    const int MT = MTOK / 128;   // 392 row tiles

    // h = LN1(x)  (f16)
    ln_kernel<<<MTOK, 256, 0, stream>>>(x, n1_g, n1_b, hbuf);

    // ---- spatial branch ----
    gemm_kernel<0><<<dim3(MT, 1536 / 64), 128, 0, stream>>>(
        hbuf, wqs, bcs, nullptr, qkvb, MTOK, 1536, 512);
    attn_kernel<NN, 224, false><<<BB * TT * HEADS, 64, 0, stream>>>(
        qkvb, aob, nullptr);
    gemm_kernel<1><<<dim3(MT, 512 / 64), 128, 0, stream>>>(
        aob, wps, s_p_b, x, xnew, MTOK, 512, 512);

    // ---- temporal branch (reuses qkv / attn-out buffers) ----
    gemm_kernel<0><<<dim3(MT, 1536 / 64), 128, 0, stream>>>(
        hbuf, wqt, bct, nullptr, qkvb, MTOK, 1536, 512);
    attn_kernel<TT, 64, true><<<BB * NN * HEADS, 64, 0, stream>>>(
        qkvb, aob, mask);
    gemm_kernel<1><<<dim3(MT, 512 / 64), 128, 0, stream>>>(
        aob, wpt, t_p_b, xnew, xnew, MTOK, 512, 512);

    // ---- FFN ----
    ln_kernel<<<MTOK, 256, 0, stream>>>(xnew, n2_g, n2_b, hbuf);
    gemm_kernel<2><<<dim3(MT, HIDDEN / 64), 128, 0, stream>>>(
        hbuf, w1h, b1, nullptr, qkvb, MTOK, HIDDEN, 512);
    gemm_kernel<3><<<dim3(MT, 512 / 64), 128, 0, stream>>>(
        qkvb, w2h, b2, xnew, d_out, MTOK, 512, HIDDEN);
}
